// SequentialMeasurementPredictor_15891378995606
// MI455X (gfx1250) — matrix operand order repaired and checked by reading, not compiled, not run on hardware
//
#include <hip/hip_runtime.h>

// ---------------------------------------------------------------------------
// CDNA5 / gfx1250 implementation of SequentialMeasurementPredictor.
//  - All GEMMs run on v_wmma_f32_16x16x32_f16 (wave32 WMMA).
//  - Each wave owns 16 samples for the full 16-step sequential chain; the
//    growing feature vector lives in LDS as f16 (A-fragment-ready layout).
//  - Weights are pre-transposed/padded to f16 [N][Kpad] in workspace so a
//    lane's B-fragment is one contiguous 32B chunk (k = 16*(lane>>4) + s, same order as A).
// ---------------------------------------------------------------------------

typedef __attribute__((ext_vector_type(16))) _Float16 v16h;
typedef __attribute__((ext_vector_type(8)))  float    v8f;

#define NSTEPS 16
#define KMAX   288                   // 17*16=272 padded to 32 -> 288
#define SPB    128                   // samples per block = 8 waves * 16 rows

struct Offsets { unsigned wo[96]; unsigned bo[96]; };   // halves / floats
struct SrcPtrs { const float* w[96]; const float* b[96]; };

// idx = m*6 + model*3 + layer   (model 0 = pred, 1 = rec)
__host__ __device__ static inline void layer_dims(int idx, int& K, int& Kp,
                                                  int& N, int& Np) {
  int m = idx / 6, sub = idx % 6, model = sub / 3, layer = sub % 3;
  K  = (layer == 0) ? 17 * (m + 1) : 256;
  Kp = (layer == 0) ? ((K + 31) & ~31) : 256;
  N  = (layer < 2) ? 256 : (model == 0 ? 8 : 32);
  Np = (layer < 2) ? 256 : (model == 0 ? 16 : 32);
}

// -------------------------- device helpers ---------------------------------

static __device__ __forceinline__ void wave_sync_lds() {
  // wave-private producer->consumer ordering through LDS (no block barrier
  // needed: each wave touches only its own rows).
  __builtin_amdgcn_wave_barrier();
  asm volatile("s_wait_dscnt 0" ::: "memory");
  __builtin_amdgcn_wave_barrier();
}

static __device__ __forceinline__ v8f wmma16(v16h a, v16h b, v8f c) {
  // (neg_a, A, neg_b, B, c_mod, C, reuse_a, reuse_b)
  return __builtin_amdgcn_wmma_f32_16x16x32_f16(false, a, false, b,
                                                (short)0, c, false, false);
}

// A fragment (16x32 f16): lane group g = lane>>4 selects K sub-pattern:
//   halves[0..7]  = X[row][k0 + 16g .. +7]
//   halves[8..15] = X[row][k0 + 16g + 8 .. +7]
static __device__ __forceinline__ v16h ld_a_frag(const _Float16* row,
                                                 int k0, int g) {
  union { float4 q[2]; v16h h; } u;
  u.q[0] = *(const float4*)(row + k0 + 16 * g);
  u.q[1] = *(const float4*)(row + k0 + 16 * g + 8);
  return u.h;
}

// Hidden layer: Y[16x256] = relu(X[16xKp] @ W + b), f16 out to LDS.
// W in global f16, layout [256][Kp] (transposed); lane's B-frag is the
// contiguous 32B chunk W[n][k0 + 16*(lane>>4) .. +15].
static __device__ void mlp_hidden(const _Float16* Xw, int xstride, int Kp,
                                  const _Float16* __restrict__ Wg,
                                  const float* __restrict__ bias,
                                  _Float16* Yw, int lane) {
  const int nsub = lane & 15;
  const int g    = lane >> 4;
  const _Float16* Xrow = Xw + (size_t)nsub * xstride;   // A row = lane&15
  __builtin_prefetch(Wg, 0, 1);

  for (int nc = 0; nc < 256; nc += 64) {
    v8f acc[4] = {};
    for (int k0 = 0; k0 < Kp; k0 += 32) {
      v16h a = ld_a_frag(Xrow, k0, g);
      const _Float16* wp = Wg + (size_t)(nc + nsub) * Kp + k0 + 16 * g;
      v16h b0 = *(const v16h*)(wp);
      v16h b1 = *(const v16h*)(wp + (size_t)16 * Kp);
      v16h b2 = *(const v16h*)(wp + (size_t)32 * Kp);
      v16h b3 = *(const v16h*)(wp + (size_t)48 * Kp);
      acc[0] = wmma16(a, b0, acc[0]);
      acc[1] = wmma16(a, b1, acc[1]);
      acc[2] = wmma16(a, b2, acc[2]);
      acc[3] = wmma16(a, b3, acc[3]);
    }
#pragma unroll
    for (int t = 0; t < 4; ++t) {
      int n = nc + t * 16 + nsub;         // C/D: col = lane&15
      float bv = bias[n];
#pragma unroll
      for (int r = 0; r < 8; ++r) {       // C/D: row = 8*(lane>>4) + r
        float y = acc[t][r] + bv;
        y = y > 0.0f ? y : 0.0f;
        Yw[(size_t)(8 * g + r) * 256 + n] = (_Float16)y;
      }
    }
  }
}

// Final layer: stg[16xNp] = X[16x256] @ W + b  (f32, no relu), Np in {16,32}.
static __device__ void mlp_final(const _Float16* Xw,
                                 const _Float16* __restrict__ Wg,
                                 const float* __restrict__ bias,
                                 float* stg, int Np, int lane) {
  const int nsub = lane & 15;
  const int g    = lane >> 4;
  const _Float16* Xrow = Xw + (size_t)nsub * 256;
  for (int nc = 0; nc < Np; nc += 16) {
    v8f acc = {};
    for (int k0 = 0; k0 < 256; k0 += 32) {
      v16h a = ld_a_frag(Xrow, k0, g);
      v16h b = *(const v16h*)(Wg + (size_t)(nc + nsub) * 256 + k0 + 16 * g);
      acc = wmma16(a, b, acc);
    }
    int n = nc + nsub;
    float bv = bias[n];
#pragma unroll
    for (int r = 0; r < 8; ++r)
      stg[(size_t)(8 * g + r) * Np + n] = acc[r] + bv;
  }
}

// ------------------------------ kernels ------------------------------------

__global__ __launch_bounds__(256)
void prep_weights_kernel(SrcPtrs P, Offsets off,
                         _Float16* __restrict__ wh, float* __restrict__ bias) {
  int idx = blockIdx.x;                 // 96 layer jobs
  int K, Kp, N, Np;
  layer_dims(idx, K, Kp, N, Np);
  const float* W  = P.w[idx];           // source: row-major (K, N)
  const float* bs = P.b[idx];
  _Float16* wd = wh + off.wo[idx];      // dest: [Np][Kp] f16 (transposed)
  float*    bd = bias + off.bo[idx];
  int total = Np * Kp;
  for (int t = threadIdx.x; t < total; t += blockDim.x) {
    int n = t / Kp, k = t - n * Kp;
    float v = (n < N && k < K) ? W[(size_t)k * N + n] : 0.0f;
    wd[(size_t)n * Kp + k] = (_Float16)v;
  }
  for (int n = threadIdx.x; n < Np; n += blockDim.x)
    bd[n] = (n < N) ? bs[n] : 0.0f;
}

__global__ __launch_bounds__(256)
void smp_main_kernel(const float* __restrict__ meas,
                     const float* __restrict__ br,
                     const float* __restrict__ bi,
                     const float* __restrict__ rho,
                     const _Float16* __restrict__ whbase,
                     const float* __restrict__ biasbase,
                     float* __restrict__ out,
                     Offsets off, int Btot) {
  extern __shared__ _Float16 smem[];
  _Float16* Xall   = smem;                              // 128 x 288 f16
  _Float16* bufall = smem + SPB * KMAX;                 // 8 waves x 2 x 16x256
  float*    stgall = (float*)(bufall + 8 * 2 * 16 * 256);  // 8 x 16x32 f32

  const int lane = threadIdx.x & 31;
  const int wid  = threadIdx.x >> 5;
  _Float16* X    = Xall + (size_t)wid * 16 * KMAX;      // this wave's 16 rows
  _Float16* bufA = bufall + (size_t)wid * 2 * 16 * 256;
  _Float16* bufB = bufA + 16 * 256;
  float*    stg  = stgall + (size_t)wid * 16 * 32;
  const int gb0  = blockIdx.x * SPB + wid * 16;         // first global sample

  // ---- initialize feature state (zero-padded to KMAX) ----
  for (int idx = lane; idx < 16 * KMAX; idx += 32) X[idx] = (_Float16)0.0f;
  if (lane < 16) {
    int b = gb0 + lane;
    _Float16* xr = X + (size_t)lane * KMAX;
    float mv = meas[b];
    xr[0] = (_Float16)mv;
#pragma unroll
    for (int k = 0; k < 8; ++k) {                        // interleave re/im
      xr[1 + 2 * k] = (_Float16)br[(size_t)b * 8 + k];
      xr[2 + 2 * k] = (_Float16)bi[(size_t)b * 8 + k];
    }
    out[(size_t)Btot * 512 + b] = mv;                    // measurement output
  }
  wave_sync_lds();

  for (int m = 0; m < NSTEPS; ++m) {
    const int K0p = ((17 * (m + 1)) + 31) & ~31;

    // ---------------- rec MLP: (i*17)->256->256->32 ----------------
    {
      int idx = m * 6 + 3;                               // model 1 = rec
      mlp_hidden(X, KMAX, K0p, whbase + off.wo[idx + 0],
                 biasbase + off.bo[idx + 0], bufA, lane);
      wave_sync_lds();
      mlp_hidden(bufA, 256, 256, whbase + off.wo[idx + 1],
                 biasbase + off.bo[idx + 1], bufB, lane);
      wave_sync_lds();
      mlp_final(bufB, whbase + off.wo[idx + 2],
                biasbase + off.bo[idx + 2], stg, 32, lane);
      wave_sync_lds();
      for (int t = lane; t < 16 * 32; t += 32) {         // recon -> HBM
        int s = t >> 5, j = t & 31;
        out[(size_t)(gb0 + s) * 512 + m * 32 + j] = stg[s * 32 + j];
      }
    }

    // ---------------- pred MLP: (i*17)->256->256->8 ----------------
    {
      int idx = m * 6;                                   // model 0 = pred
      mlp_hidden(X, KMAX, K0p, whbase + off.wo[idx + 0],
                 biasbase + off.bo[idx + 0], bufA, lane);
      wave_sync_lds();
      mlp_hidden(bufA, 256, 256, whbase + off.wo[idx + 1],
                 biasbase + off.bo[idx + 1], bufB, lane);
      wave_sync_lds();
      mlp_final(bufB, whbase + off.wo[idx + 2],
                biasbase + off.bo[idx + 2], stg, 16, lane);
      wave_sync_lds();
    }

    // -------- per-sample projector build + rho contraction --------
    if (lane < 16) {
      int s = lane, b = gb0 + s;
      float p[8];
#pragma unroll
      for (int j = 0; j < 8; ++j) p[j] = stg[s * 16 + j];

      float Mre[2][2][2], Mim[2][2][2];
#pragma unroll
      for (int q = 0; q < 2; ++q) {
        float ar = p[q * 4 + 0], ai = p[q * 4 + 1];      // v0
        float b0r = p[q * 4 + 2], b0i = p[q * 4 + 3];    // v1
        float n0 = ar * ar + ai * ai;
        float n1 = b0r * b0r + b0i * b0i;
        float cr = ar * b0r + ai * b0i;                  // v0*conj(v1)
        float ci = ai * b0r - ar * b0i;
        float inv = 1.0f / (n0 + n1);
        Mre[q][0][0] = n0 * inv;  Mim[q][0][0] = 0.0f;
        Mre[q][0][1] = cr * inv;  Mim[q][0][1] = ci * inv;
        Mre[q][1][0] = cr * inv;  Mim[q][1][0] = -ci * inv;
        Mre[q][1][1] = n1 * inv;  Mim[q][1][1] = 0.0f;
      }
      // new_meas = Re sum M0[x][y]*M1[u][v]*rho_c[y][v][x][u]
      float nm = 0.0f;
#pragma unroll
      for (int x = 0; x < 2; ++x)
#pragma unroll
        for (int y = 0; y < 2; ++y)
#pragma unroll
          for (int u = 0; u < 2; ++u)
#pragma unroll
            for (int v = 0; v < 2; ++v) {
              float tre = Mre[0][x][y] * Mre[1][u][v] - Mim[0][x][y] * Mim[1][u][v];
              float tim = Mre[0][x][y] * Mim[1][u][v] + Mim[0][x][y] * Mre[1][u][v];
              int r = 2 * y + v, c = 2 * x + u;
              float rr = rho[(size_t)b * 32 + r * 4 + c];
              float ri = rho[(size_t)b * 32 + 16 + r * 4 + c];
              nm += tre * rr - tim * ri;
            }
      // append 17 new features
      _Float16* xr = X + (size_t)s * KMAX + 17 * (m + 1);
      xr[0] = (_Float16)nm;
#pragma unroll
      for (int q = 0; q < 2; ++q)
#pragma unroll
        for (int r = 0; r < 2; ++r)
#pragma unroll
          for (int c = 0; c < 2; ++c) {
            int k = (q * 2 + r) * 2 + c;
            xr[1 + 2 * k] = (_Float16)Mre[q][r][c];
            xr[2 + 2 * k] = (_Float16)Mim[q][r][c];
          }
    }
    wave_sync_lds();
  }
}

// ------------------------------ launcher -----------------------------------

extern "C" void kernel_launch(void* const* d_in, const int* in_sizes, int n_in,
                              void* d_out, int out_size, void* d_ws,
                              size_t ws_size, hipStream_t stream) {
  (void)n_in; (void)out_size; (void)ws_size;
  const int Btot = in_sizes[0];                          // B (measurement B x 1)

  Offsets off;
  size_t wcur = 0, bcur = 0;
  for (int idx = 0; idx < 96; ++idx) {
    int K, Kp, N, Np;
    layer_dims(idx, K, Kp, N, Np);
    off.wo[idx] = (unsigned)wcur;
    off.bo[idx] = (unsigned)bcur;
    wcur += (size_t)Np * Kp;
    bcur += (size_t)Np;
  }
  _Float16* wh   = (_Float16*)d_ws;
  float*    bias = (float*)((char*)d_ws +
                            ((wcur * sizeof(_Float16) + 255) & ~(size_t)255));

  SrcPtrs P;
  for (int idx = 0; idx < 96; ++idx) {
    int m = idx / 6, sub = idx % 6, model = sub / 3, layer = sub % 3;
    int base = 4 + model * 96 + m * 6 + layer * 2;       // pred block, then rec
    P.w[idx] = (const float*)d_in[base];
    P.b[idx] = (const float*)d_in[base + 1];
  }

  hipLaunchKernelGGL(prep_weights_kernel, dim3(96), dim3(256), 0, stream,
                     P, off, wh, bias);

  const size_t shbytes = (size_t)(SPB * KMAX + 8 * 2 * 16 * 256) * 2 +
                         (size_t)(8 * 16 * 32) * 4;      // 221,184 B
  hipLaunchKernelGGL(smp_main_kernel, dim3(Btot / SPB), dim3(256), shbytes,
                     stream,
                     (const float*)d_in[0], (const float*)d_in[1],
                     (const float*)d_in[2], (const float*)d_in[3],
                     (const _Float16*)wh, (const float*)bias,
                     (float*)d_out, off, Btot);
}
